// CrossImageAttentionLayer_76759655514671
// MI455X (gfx1250) — compile-verified, matrix operand order checked by reading, not run on hardware
//
#include <hip/hip_runtime.h>
#include <hip/hip_bf16.h>

typedef __attribute__((ext_vector_type(16))) __bf16 bf16x16;
typedef __attribute__((ext_vector_type(8)))  __bf16 bf16x8;
typedef __attribute__((ext_vector_type(8)))  float  f32x8;
typedef __attribute__((ext_vector_type(4)))  float  f32x4;

namespace {
constexpr int kD = 256;   // embedding dim
constexpr int kC = 8;     // cameras
constexpr int kH = 64;
constexpr int kW = 64;
constexpr int kTile = 16; // query rows per block
constexpr int kB = 2;     // batch
constexpr int kKSteps = kD / 32;  // 8 WMMA k-steps (16x16x32)
}

// ---------------- weight prep: fp32 row-major -> bf16 transposed [n][k] ----
__global__ void prep_weights(const float* __restrict__ Wq,
                             const float* __restrict__ Wkv,
                             const float* __restrict__ Wo,
                             __bf16* __restrict__ WqT,
                             __bf16* __restrict__ WkvT,
                             __bf16* __restrict__ WoT) {
  const int idx = blockIdx.x * blockDim.x + threadIdx.x;
  if (idx < kD * kD) {
    const int n = idx / kD, k = idx % kD;
    WqT[idx] = (__bf16)Wq[k * kD + n];
  } else if (idx < kD * kD + kD * 2 * kD) {
    const int e = idx - kD * kD;
    const int n = e / kD, k = e % kD;            // n in [0,512)
    WkvT[e] = (__bf16)Wkv[k * (2 * kD) + n];
  } else if (idx < 2 * kD * kD + kD * 2 * kD) {
    const int e = idx - (kD * kD + kD * 2 * kD);
    const int n = e / kD, k = e % kD;
    WoT[e] = (__bf16)Wo[k * kD + n];
  }
}

// ------------- WMMA fragment loaders (16-bit A 16x32 layout, B mirrored) ---
__device__ inline bf16x16 load_a_frag(const __bf16 (*A)[kD], int lane, int ks) {
  const int m  = lane & 15;
  const int kb = ks * 32 + ((lane >> 4) << 3);
  bf16x8 lo = *(const bf16x8*)&A[m][kb];
  bf16x8 hi = *(const bf16x8*)&A[m][kb + 16];
  bf16x16 f;
#pragma unroll
  for (int i = 0; i < 8; ++i) { f[i] = lo[i]; f[i + 8] = hi[i]; }
  return f;
}

__device__ inline bf16x16 load_b_frag(const __bf16* __restrict__ WT, int lane,
                                      int jt, int ks) {
  const int n  = (jt << 4) + (lane & 15);
  const int kb = ks * 32 + ((lane >> 4) << 3);
  const __bf16* p = WT + (size_t)n * kD + kb;
  __builtin_prefetch(p + 32, 0, 1);
  bf16x8 lo = *(const bf16x8*)p;
  bf16x8 hi = *(const bf16x8*)(p + 16);
  bf16x16 f;
#pragma unroll
  for (int i = 0; i < 8; ++i) { f[i] = lo[i]; f[i + 8] = hi[i]; }
  return f;
}

__device__ inline f32x8 zero8() {
  return (f32x8){0.f, 0.f, 0.f, 0.f, 0.f, 0.f, 0.f, 0.f};
}

// Two 16x16 output tiles (jt0, jt1) of A(16x256 LDS bf16) x W^T(global bf16).
// Shared A fragment, two independent accumulator chains (no WMMA RAW hazard).
__device__ inline void gemm_tile_pair(const __bf16 (*A)[kD],
                                      const __bf16* __restrict__ WT,
                                      int lane, int jt0, int jt1,
                                      f32x8& acc0, f32x8& acc1) {
  acc0 = zero8();
  acc1 = zero8();
#pragma unroll
  for (int ks = 0; ks < kKSteps; ++ks) {
    const bf16x16 a  = load_a_frag(A, lane, ks);
    const bf16x16 b0 = load_b_frag(WT, lane, jt0, ks);
    const bf16x16 b1 = load_b_frag(WT, lane, jt1, ks);
    acc0 = __builtin_amdgcn_wmma_f32_16x16x32_bf16(false, a, false, b0,
                                                   (short)0, acc0, false, false);
    acc1 = __builtin_amdgcn_wmma_f32_16x16x32_bf16(false, a, false, b1,
                                                   (short)0, acc1, false, false);
  }
}

// ---------------------------- fused main kernel ---------------------------
__global__ __launch_bounds__(256)
void fused_cross_attn(const float* __restrict__ queries,
                      const float* __restrict__ feat,
                      const float* __restrict__ coords,
                      const int*   __restrict__ vmask,
                      const float* __restrict__ bq,
                      const float* __restrict__ bkv,
                      const float* __restrict__ bo,
                      const float* __restrict__ gamma,
                      const float* __restrict__ beta,
                      const __bf16* __restrict__ WqT,
                      const __bf16* __restrict__ WkvT,
                      const __bf16* __restrict__ WoT,
                      float* __restrict__ out, int N) {
  __shared__ __bf16 qn_s[kTile][kD];          // LN(q) bf16; reused for "mix"
  __shared__ __bf16 samp_s[kC][kTile][kD];    // bilinear samples per camera
  __shared__ __bf16 vm_s[kTile][kD];          // mix @ Wv (bf16)
  __shared__ float  q_s[kTile][kD];           // q projection fp32
  __shared__ float  scp_s[kC][kTile][16];     // score partials per jt
  __shared__ float  sc_s[kTile][kC];
  __shared__ float  attn_s[kTile][kC];
  __shared__ float  asum_s[kTile];
  __shared__ int    mv_s[kTile][kC];

  const int b    = blockIdx.y;
  const int n0   = blockIdx.x * kTile;
  const int tid  = threadIdx.x;
  const int wave = tid >> 5;                  // 8 waves (wave32)
  const int lane = tid & 31;

  // ---- 1. LayerNorm: wave w handles rows 2w, 2w+1 ----
#pragma unroll
  for (int rr = 0; rr < 2; ++rr) {
    const int r = wave * 2 + rr;
    const float* qrow = queries + ((size_t)b * N + n0 + r) * kD;
    float x[8];
    float s = 0.f;
#pragma unroll
    for (int i = 0; i < 8; ++i) { x[i] = qrow[lane + 32 * i]; s += x[i]; }
#pragma unroll
    for (int m = 16; m >= 1; m >>= 1) s += __shfl_xor(s, m, 32);
    const float mu = s * (1.f / kD);
    float vs = 0.f;
#pragma unroll
    for (int i = 0; i < 8; ++i) { const float dd = x[i] - mu; vs += dd * dd; }
#pragma unroll
    for (int m = 16; m >= 1; m >>= 1) vs += __shfl_xor(vs, m, 32);
    const float rstd = rsqrtf(vs * (1.f / kD) + 1e-5f);
#pragma unroll
    for (int i = 0; i < 8; ++i) {
      const int d = lane + 32 * i;
      qn_s[r][d] = (__bf16)((x[i] - mu) * rstd * gamma[d] + beta[d]);
    }
  }

  // ---- 2. bilinear sampling: wave == camera ----
  {
    const int c = wave;
    const float* fbase = feat + ((size_t)b * kC + c) * (size_t)kH * kW * kD;
    for (int r = 0; r < kTile; ++r) {
      const size_t cn = (((size_t)b * kC + c) * N + (size_t)(n0 + r));
      const float cx = coords[cn * 2 + 0];
      const float cy = coords[cn * 2 + 1];
      const float fx = (cx + 1.f) * 0.5f * (kW - 1);
      const float fy = (cy + 1.f) * 0.5f * (kH - 1);
      const float x0f = floorf(fx), y0f = floorf(fy);
      const int   x0 = (int)x0f,   y0 = (int)y0f;
      const float wx1 = fx - x0f,  wy1 = fy - y0f;
      const float wgt[4] = {(1.f - wy1) * (1.f - wx1), (1.f - wy1) * wx1,
                            wy1 * (1.f - wx1),         wy1 * wx1};
      const int xs[4] = {x0, x0 + 1, x0, x0 + 1};
      const int ys[4] = {y0, y0, y0 + 1, y0 + 1};
      float acc[8] = {0.f, 0.f, 0.f, 0.f, 0.f, 0.f, 0.f, 0.f};
#pragma unroll
      for (int cor = 0; cor < 4; ++cor) {
        const bool inb = xs[cor] >= 0 && xs[cor] < kW &&
                         ys[cor] >= 0 && ys[cor] < kH;
        const int xc = min(max(xs[cor], 0), kW - 1);
        const int yc = min(max(ys[cor], 0), kH - 1);
        const float w = inb ? wgt[cor] : 0.f;
        const float* p = fbase + ((size_t)yc * kW + xc) * kD + lane * 8;
        const f32x4 v0 = *(const f32x4*)p;
        const f32x4 v1 = *(const f32x4*)(p + 4);
#pragma unroll
        for (int i = 0; i < 4; ++i) { acc[i] += w * v0[i]; acc[4 + i] += w * v1[i]; }
      }
#pragma unroll
      for (int i = 0; i < 8; ++i) samp_s[c][r][lane * 8 + i] = (__bf16)acc[i];
    }
    if (lane < kTile)
      mv_s[lane][c] = vmask[((size_t)b * kC + c) * N + n0 + lane];
  }
  __syncthreads();

  // ---- 3. q = LN(q) @ Wq + bq (WMMA, paired tiles) ----
  {
    f32x8 a0, a1;
    gemm_tile_pair(qn_s, WqT, lane, wave, wave + 8, a0, a1);
    const int M0 = (lane >> 4) << 3;
    const int c0 = (wave << 4) + (lane & 15);
    const int c1 = ((wave + 8) << 4) + (lane & 15);
#pragma unroll
    for (int r = 0; r < 8; ++r) {
      q_s[M0 + r][c0] = a0[r] + bq[c0];
      q_s[M0 + r][c1] = a1[r] + bq[c1];
    }
  }
  __syncthreads();

  // ---- 4. k projection for ALL cameras per column tile (B-frag reuse,
  //         8 independent WMMA chains), scores folded from accumulators ----
#pragma unroll
  for (int t = 0; t < 2; ++t) {
    const int jt = wave + t * 8;
    f32x8 acc[kC];
#pragma unroll
    for (int c = 0; c < kC; ++c) acc[c] = zero8();
#pragma unroll
    for (int ks = 0; ks < kKSteps; ++ks) {
      const bf16x16 bfrag = load_b_frag(WkvT, lane, jt, ks);
#pragma unroll
      for (int c = 0; c < kC; ++c) {
        const bf16x16 afrag = load_a_frag(samp_s[c], lane, ks);
        acc[c] = __builtin_amdgcn_wmma_f32_16x16x32_bf16(
            false, afrag, false, bfrag, (short)0, acc[c], false, false);
      }
    }
    // fold scores: partial[c][M] += sum_n q[M][n] * (k_acc + bk[n])
    const int M0 = (lane >> 4) << 3;
    const int n  = (jt << 4) + (lane & 15);
    const float bk = bkv[n];
    float qv[8];
#pragma unroll
    for (int r = 0; r < 8; ++r) qv[r] = q_s[M0 + r][n];
#pragma unroll
    for (int c = 0; c < kC; ++c) {
      float tt[8];
#pragma unroll
      for (int r = 0; r < 8; ++r) tt[r] = (acc[c][r] + bk) * qv[r];
#pragma unroll
      for (int m = 8; m >= 1; m >>= 1) {
#pragma unroll
        for (int r = 0; r < 8; ++r) tt[r] += __shfl_xor(tt[r], m, 32);
      }
      if ((lane & 15) == 0) {
#pragma unroll
        for (int r = 0; r < 8; ++r) scp_s[c][M0 + r][jt] = tt[r];
      }
    }
  }
  __syncthreads();

  // ---- 5. fixed-order score reduction over 16 column tiles ----
  if (tid < kC * kTile) {
    const int c = tid >> 4, r = tid & 15;
    float s = 0.f;
#pragma unroll
    for (int j = 0; j < 16; ++j) s += scp_s[c][r][j];
    sc_s[r][c] = s;
  }
  __syncthreads();

  // ---- 6. masked softmax over C=8 (NaN -> 0 when fully masked) ----
  if (tid < kTile) {
    const int r = tid;
    float mx = -3.0e38f;
    int any = 0;
    float sc[kC];
#pragma unroll
    for (int c = 0; c < kC; ++c) {
      sc[c] = sc_s[r][c] * 0.0625f;           // 1/sqrt(256)
      if (mv_s[r][c]) { any = 1; mx = fmaxf(mx, sc[c]); }
    }
    float sum = 0.f, e[kC];
#pragma unroll
    for (int c = 0; c < kC; ++c) {
      e[c] = mv_s[r][c] ? __expf(sc[c] - mx) : 0.f;
      sum += e[c];
    }
    const float inv = (any && sum > 0.f) ? 1.f / sum : 0.f;
#pragma unroll
    for (int c = 0; c < kC; ++c) attn_s[r][c] = e[c] * inv;
    asum_s[r] = sum * inv;                    // 1 if any valid else 0 (for bv)
  }
  __syncthreads();

  // ---- 7. mix = sum_c attn_c * sampled_c (attn applied pre-projection) ----
  {
    const int r = tid >> 4, seg = tid & 15;
#pragma unroll
    for (int j = 0; j < 16; ++j) {
      const int d = seg * 16 + j;
      float m = 0.f;
#pragma unroll
      for (int c = 0; c < kC; ++c) m += attn_s[r][c] * (float)samp_s[c][r][d];
      qn_s[r][d] = (__bf16)m;                 // reuse qn_s as mix
    }
  }
  __syncthreads();

  // ---- 8. v_mix = mix @ Wv + asum*bv (WMMA, paired tiles) ----
  {
    f32x8 a0, a1;
    gemm_tile_pair(qn_s, WkvT + (size_t)kD * kD, lane, wave, wave + 8, a0, a1);
    const int M0 = (lane >> 4) << 3;
    const int c0 = (wave << 4) + (lane & 15);
    const int c1 = ((wave + 8) << 4) + (lane & 15);
#pragma unroll
    for (int r = 0; r < 8; ++r) {
      vm_s[M0 + r][c0] = (__bf16)(a0[r] + asum_s[M0 + r] * bkv[kD + c0]);
      vm_s[M0 + r][c1] = (__bf16)(a1[r] + asum_s[M0 + r] * bkv[kD + c1]);
    }
  }
  __syncthreads();

  // ---- 9. out = residual + v_mix @ Wo + bo (WMMA, paired tiles) ----
  {
    f32x8 a0, a1;
    gemm_tile_pair(vm_s, WoT, lane, wave, wave + 8, a0, a1);
    const int M0 = (lane >> 4) << 3;
    const int c0 = (wave << 4) + (lane & 15);
    const int c1 = ((wave + 8) << 4) + (lane & 15);
#pragma unroll
    for (int r = 0; r < 8; ++r) {
      const size_t o0 = ((size_t)b * N + n0 + M0 + r) * kD + c0;
      const size_t o1 = ((size_t)b * N + n0 + M0 + r) * kD + c1;
      out[o0] = queries[o0] + a0[r] + bo[c0];
      out[o1] = queries[o1] + a1[r] + bo[c1];
    }
  }
}

// ------------------------------- launcher ---------------------------------
extern "C" void kernel_launch(void* const* d_in, const int* in_sizes, int n_in,
                              void* d_out, int out_size, void* d_ws,
                              size_t ws_size, hipStream_t stream) {
  const float* queries = (const float*)d_in[0];
  const float* feat    = (const float*)d_in[1];
  const float* coords  = (const float*)d_in[2];
  const int*   vmask   = (const int*)d_in[3];
  const float* Wq      = (const float*)d_in[4];
  const float* bq      = (const float*)d_in[5];
  const float* Wkv     = (const float*)d_in[6];
  const float* bkv     = (const float*)d_in[7];
  const float* Wo      = (const float*)d_in[8];
  const float* bo      = (const float*)d_in[9];
  const float* gamma   = (const float*)d_in[10];
  const float* beta    = (const float*)d_in[11];
  float* out = (float*)d_out;

  __bf16* WqT  = (__bf16*)d_ws;                 // 256x256
  __bf16* WkvT = WqT + kD * kD;                 // 512x256
  __bf16* WoT  = WkvT + kD * 2 * kD;            // 256x256

  const int total = 2 * kD * kD + kD * 2 * kD;  // 262144 elements
  prep_weights<<<(total + 255) / 256, 256, 0, stream>>>(Wq, Wkv, Wo,
                                                        WqT, WkvT, WoT);

  const int N = in_sizes[0] / (kB * kD);        // 16384
  fused_cross_attn<<<dim3(N / kTile, kB), 256, 0, stream>>>(
      queries, feat, coords, vmask, bq, bkv, bo, gamma, beta,
      WqT, WkvT, WoT, out, N);
}
